// GQAAttention_82910048682137
// MI455X (gfx1250) — matrix operand order repaired and checked by reading, not compiled, not run on hardware
//
#include <hip/hip_runtime.h>
#include <hip/hip_bf16.h>

#define B_   2
#define S_   2048
#define DM_  2048
#define H_   32
#define G_   8
#define HD_  64
#define QDIM 2048
#define KVDIM 512
#define NQKV 3072

typedef __attribute__((ext_vector_type(16))) __bf16 bf16x16;
typedef __attribute__((ext_vector_type(8)))  __bf16 bf16x8;
typedef __attribute__((ext_vector_type(8)))  float  f32x8;

// A-matrix fragment (16-bit 16x32): per-lane two 16B chunks at +0 and +16 elems.
__device__ __forceinline__ bf16x16 fragA_ld(const __bf16* lo) {
  bf16x8 a = *(const bf16x8*)lo;
  bf16x8 b = *(const bf16x8*)(lo + 16);
  return __builtin_shufflevector(a, b, 0,1,2,3,4,5,6,7,8,9,10,11,12,13,14,15);
}

// B-matrix fragment (16-bit 32x16): per-lane two 16B chunks at +0 and +16 elems.
__device__ __forceinline__ bf16x16 fragB_ld(const __bf16* p) {
  bf16x8 a = *(const bf16x8*)p;
  bf16x8 b = *(const bf16x8*)(p + 16);
  return __builtin_shufflevector(a, b, 0,1,2,3,4,5,6,7,8,9,10,11,12,13,14,15);
}

__device__ __forceinline__ f32x8 wmma_bf16(bf16x16 a, bf16x16 b, f32x8 c) {
  return __builtin_amdgcn_wmma_f32_16x16x32_bf16(false, a, false, b, (short)0, c,
                                                 false, false);
}

// ---------------------------------------------------------------- conversion
__global__ __launch_bounds__(256) void f32_to_bf16_kernel(
    const float* __restrict__ src, __bf16* __restrict__ dst, size_t n) {
  size_t i = (size_t)blockIdx.x * blockDim.x + threadIdx.x;
  size_t stride = (size_t)gridDim.x * blockDim.x;
  for (; i < n; i += stride) dst[i] = (__bf16)src[i];
}

// ----------------------------------------------------- 16x64-per-wave GEMM core
// C[M,N] += A[M,K] (row major) x W[N,K]^T (row major, K contiguous).
// Explicit ping-pong double buffering (unroll-2, no register rotation copies):
// loads for the next k-step fly while WMMAs for the current step execute.
__device__ __forceinline__ void gemm_tile_16x64(
    const __bf16* __restrict__ A, const __bf16* __restrict__ W,
    int K, int arow, int bcol, f32x8 acc[4]) {
  const int lane = threadIdx.x & 31;
  const int lm   = lane & 15;
  const int kbA  = (lane >> 4) * 8;
  const int kbB  = (lane >> 4) * 8;
  const __bf16* ap = A + (size_t)(arow + lm) * K + kbA;
  const __bf16* wp = W + (size_t)(bcol + lm) * K + kbB;

  bf16x16 af0 = fragA_ld(ap);
  bf16x16 wf0[4];
#pragma unroll
  for (int nt = 0; nt < 4; ++nt) wf0[nt] = fragB_ld(wp + (size_t)(nt * 16) * K);

  bf16x16 af1, wf1[4];
  int k = 0;
  for (; k < K - 64; k += 64) {
    // load k+32 into buffer 1
    af1 = fragA_ld(ap + k + 32);
#pragma unroll
    for (int nt = 0; nt < 4; ++nt)
      wf1[nt] = fragB_ld(wp + (size_t)(nt * 16) * K + k + 32);
    __builtin_prefetch(ap + k + 1024, 0, 1);
    // compute buffer 0 (k)
#pragma unroll
    for (int nt = 0; nt < 4; ++nt) acc[nt] = wmma_bf16(af0, wf0[nt], acc[nt]);
    // load k+64 into buffer 0
    af0 = fragA_ld(ap + k + 64);
#pragma unroll
    for (int nt = 0; nt < 4; ++nt)
      wf0[nt] = fragB_ld(wp + (size_t)(nt * 16) * K + k + 64);
    // compute buffer 1 (k+32)
#pragma unroll
    for (int nt = 0; nt < 4; ++nt) acc[nt] = wmma_bf16(af1, wf1[nt], acc[nt]);
  }
  // tail pair: k == K - 64
  af1 = fragA_ld(ap + k + 32);
#pragma unroll
  for (int nt = 0; nt < 4; ++nt)
    wf1[nt] = fragB_ld(wp + (size_t)(nt * 16) * K + k + 32);
#pragma unroll
  for (int nt = 0; nt < 4; ++nt) acc[nt] = wmma_bf16(af0, wf0[nt], acc[nt]);
#pragma unroll
  for (int nt = 0; nt < 4; ++nt) acc[nt] = wmma_bf16(af1, wf1[nt], acc[nt]);
}

// ----------------------------------------------------------------- QKV GEMM
// qkv[4096,3072] = Xb[4096,2048] @ Wqkv[3072,2048]^T, epilogue scatters into
// Qp[b,h,s,d] (scaled 1/8), Kp[b,g,s,d], Vt[b,g,d,s].
__global__ __launch_bounds__(256) void qkv_gemm_kernel(
    const __bf16* __restrict__ X, const __bf16* __restrict__ Wq,
    __bf16* __restrict__ Qp, __bf16* __restrict__ Kp, __bf16* __restrict__ Vt) {
  const int wave = threadIdx.x >> 5;
  const int arow = blockIdx.y * 32 + (wave >> 2) * 16;
  const int bcol = blockIdx.x * 256 + (wave & 3) * 64;
  f32x8 acc[4] = {};
  gemm_tile_16x64(X, Wq, DM_, arow, bcol, acc);

  const int lane = threadIdx.x & 31;
  const int lm = lane & 15, hb = lane >> 4;
#pragma unroll
  for (int nt = 0; nt < 4; ++nt) {
    const int col = bcol + nt * 16 + lm;
#pragma unroll
    for (int r = 0; r < 8; ++r) {
      const int row = arow + r + hb * 8;
      const int bb = row >> 11, s = row & (S_ - 1);
      const float f = acc[nt][r];
      if (col < QDIM) {
        const int h = col >> 6, d = col & 63;
        Qp[((((size_t)bb * H_ + h) * S_ + s) << 6) + d] = (__bf16)(f * 0.125f);
      } else if (col < QDIM + KVDIM) {
        const int c2 = col - QDIM, g = c2 >> 6, d = c2 & 63;
        Kp[((((size_t)bb * G_ + g) * S_ + s) << 6) + d] = (__bf16)f;
      } else {
        const int c2 = col - QDIM - KVDIM, g = c2 >> 6, d = c2 & 63;
        Vt[(((size_t)bb * G_ + g) * HD_ + d) * S_ + s] = (__bf16)f;
      }
    }
  }
}

// -------------------------------------------------------------- attention
// One wave handles a 16-row Q tile for one (b,h); flash-attention over kv
// in steps of 32, causal. P transposed C->A layout through per-wave LDS.
// V fragments are loaded before the softmax so their latency hides under it.
__global__ __launch_bounds__(128) void attn_kernel(
    const __bf16* __restrict__ Qp, const __bf16* __restrict__ Kp,
    const __bf16* __restrict__ Vt, __bf16* __restrict__ Op) {
  __shared__ __align__(16) __bf16 pbuf[4][16 * 32];
  const int wv   = threadIdx.x >> 5;
  const int lane = threadIdx.x & 31;
  const int lm   = lane & 15;
  const int hb   = lane >> 4;
  const int kbA  = hb * 8;
  const int kbB  = hb * 8;

  const int tile = blockIdx.x * 4 + wv;        // 0 .. 8191
  const int qt = tile & (S_ / 16 - 1);         // q tile index
  const int h  = (tile >> 7) & (H_ - 1);
  const int b  = tile >> 12;
  const int g  = h >> 2;                       // HPG = 4

  const __bf16* qrow = Qp + ((((size_t)b * H_ + h) * S_) + qt * 16 + lm) * HD_;
  const bf16x16 aq0 = fragA_ld(qrow + kbA);
  const bf16x16 aq1 = fragA_ld(qrow + 32 + kbA);
  const __bf16* Kb = Kp + (((size_t)b * G_ + g) * S_) * HD_;
  const __bf16* Vb = Vt + (((size_t)b * G_ + g) * HD_) * S_;

  f32x8 oacc[4] = {};
  float mrun[8], lrun[8];
#pragma unroll
  for (int r = 0; r < 8; ++r) { mrun[r] = -3.0e38f; lrun[r] = 0.0f; }

  __bf16* pw = &pbuf[wv][0];
  const int kvend = qt * 16 + 16;
  for (int kv = 0; kv < kvend; kv += 32) {
    // ---- K fragments + early V fragments (independent of softmax) ----
    const __bf16* kr0 = Kb + (size_t)(kv + lm) * HD_;
    const __bf16* kr1 = kr0 + 16 * HD_;
    const bf16x16 bk00 = fragB_ld(kr0 + kbB);
    const bf16x16 bk01 = fragB_ld(kr0 + 32 + kbB);
    const bf16x16 bk10 = fragB_ld(kr1 + kbB);
    const bf16x16 bk11 = fragB_ld(kr1 + 32 + kbB);
    bf16x16 vf[4];
#pragma unroll
    for (int c = 0; c < 4; ++c)
      vf[c] = fragB_ld(Vb + (size_t)(c * 16 + lm) * S_ + kv + kbB);
    __builtin_prefetch(kr0 + 32 * HD_, 0, 1);   // next-step K rows

    // ---- scores S = Q @ K^T (pre-scaled), two 16x16 tiles ----
    f32x8 z = {};
    f32x8 s0 = wmma_bf16(aq0, bk00, z);
    s0 = wmma_bf16(aq1, bk01, s0);
    f32x8 s1 = wmma_bf16(aq0, bk10, z);
    s1 = wmma_bf16(aq1, bk11, s1);

    // ---- causal mask + online softmax update ----
    const int c0 = kv + lm, c1 = kv + 16 + lm;
#pragma unroll
    for (int r = 0; r < 8; ++r) {
      const int qg = qt * 16 + r + hb * 8;
      const float v0 = (c0 <= qg) ? s0[r] : -3.0e38f;
      const float v1 = (c1 <= qg) ? s1[r] : -3.0e38f;
      float t = fmaxf(v0, v1);
      t = fmaxf(t, __shfl_xor(t, 1, 32));
      t = fmaxf(t, __shfl_xor(t, 2, 32));
      t = fmaxf(t, __shfl_xor(t, 4, 32));
      t = fmaxf(t, __shfl_xor(t, 8, 32));
      const float mn = fmaxf(mrun[r], t);
      const float al = __expf(mrun[r] - mn);
      const float p0 = __expf(v0 - mn);
      const float p1 = __expf(v1 - mn);
      float rs = p0 + p1;
      rs += __shfl_xor(rs, 1, 32);
      rs += __shfl_xor(rs, 2, 32);
      rs += __shfl_xor(rs, 4, 32);
      rs += __shfl_xor(rs, 8, 32);
      lrun[r] = lrun[r] * al + rs;
      mrun[r] = mn;
      oacc[0][r] *= al; oacc[1][r] *= al; oacc[2][r] *= al; oacc[3][r] *= al;
      const int prow = r + hb * 8;                 // C layout -> LDS row major
      pw[prow * 32 + lm]      = (__bf16)p0;
      pw[prow * 32 + 16 + lm] = (__bf16)p1;
    }
    __builtin_amdgcn_wave_barrier();               // same-wave LDS exchange

    // ---- O += P @ V  (P re-read in A layout via LDS) ----
    const bf16x16 pa = fragA_ld(&pw[lm * 32] + kbA);
#pragma unroll
    for (int c = 0; c < 4; ++c) oacc[c] = wmma_bf16(pa, vf[c], oacc[c]);
    __builtin_amdgcn_wave_barrier();
  }

  // ---- normalize rows and store O as [b, s, h*64 + d] bf16 ----
  float inv[8];
#pragma unroll
  for (int r = 0; r < 8; ++r) inv[r] = 1.0f / lrun[r];
#pragma unroll
  for (int c = 0; c < 4; ++c) {
#pragma unroll
    for (int r = 0; r < 8; ++r) {
      const int srow = qt * 16 + r + hb * 8;
      Op[((size_t)b * S_ + srow) * DM_ + h * HD_ + c * 16 + lm] =
          (__bf16)(oacc[c][r] * inv[r]);
    }
  }
}

// ----------------------------------------------------------------- out GEMM
__global__ __launch_bounds__(256) void out_gemm_kernel(
    const __bf16* __restrict__ O, const __bf16* __restrict__ Wo,
    float* __restrict__ out) {
  const int wave = threadIdx.x >> 5;
  const int arow = blockIdx.y * 32 + (wave >> 2) * 16;
  const int bcol = blockIdx.x * 256 + (wave & 3) * 64;
  f32x8 acc[4] = {};
  gemm_tile_16x64(O, Wo, DM_, arow, bcol, acc);

  const int lane = threadIdx.x & 31;
  const int lm = lane & 15, hb = lane >> 4;
#pragma unroll
  for (int nt = 0; nt < 4; ++nt) {
    const int col = bcol + nt * 16 + lm;
#pragma unroll
    for (int r = 0; r < 8; ++r) {
      const int row = arow + r + hb * 8;
      out[(size_t)row * DM_ + col] = acc[nt][r];
    }
  }
}

// ------------------------------------------------------------------- launch
extern "C" void kernel_launch(void* const* d_in, const int* in_sizes, int n_in,
                              void* d_out, int out_size, void* d_ws, size_t ws_size,
                              hipStream_t stream) {
  (void)in_sizes; (void)n_in; (void)out_size; (void)ws_size;
  const float* x    = (const float*)d_in[0];
  const float* wqkv = (const float*)d_in[1];
  const float* wo   = (const float*)d_in[2];
  // d_in[3] = causal mask, implicit in the kernel
  float* out = (float*)d_out;

  char* ws = (char*)d_ws;
  size_t off = 0;
  auto carve = [&](size_t bytes) -> void* {
    void* p = ws + off;
    off += (bytes + 255) & ~(size_t)255;
    return p;
  };
  const size_t NT = (size_t)B_ * S_;  // 4096 tokens
  __bf16* Xb  = (__bf16*)carve(NT * DM_ * 2);            // reused as Op later
  __bf16* Wqb = (__bf16*)carve((size_t)NQKV * DM_ * 2);
  __bf16* Wob = (__bf16*)carve((size_t)DM_ * QDIM * 2);
  __bf16* Qp  = (__bf16*)carve((size_t)B_ * H_ * S_ * HD_ * 2);
  __bf16* Kp  = (__bf16*)carve((size_t)B_ * G_ * S_ * HD_ * 2);
  __bf16* Vt  = (__bf16*)carve((size_t)B_ * G_ * S_ * HD_ * 2);
  __bf16* Op  = Xb;  // X no longer needed once QKV GEMM is done (same stream)

  f32_to_bf16_kernel<<<dim3(2048), 256, 0, stream>>>(x, Xb, NT * (size_t)DM_);
  f32_to_bf16_kernel<<<dim3(2048), 256, 0, stream>>>(wqkv, Wqb, (size_t)NQKV * DM_);
  f32_to_bf16_kernel<<<dim3(1024), 256, 0, stream>>>(wo, Wob, (size_t)DM_ * QDIM);

  qkv_gemm_kernel<<<dim3(NQKV / 256, NT / 32), 256, 0, stream>>>(Xb, Wqb, Qp, Kp, Vt);
  attn_kernel<<<dim3((B_ * H_ * (S_ / 16)) / 4), 128, 0, stream>>>(Qp, Kp, Vt, Op);
  out_gemm_kernel<<<dim3(DM_ / 256, NT / 32), 256, 0, stream>>>(Op, Wob, out);
}
